// Upsample_MLP_16836271800459
// MI455X (gfx1250) — matrix operand order repaired and checked by reading, not compiled, not run on hardware
//
#include <hip/hip_runtime.h>

#define B_   8
#define H_   256
#define W_   256
#define HH   128
#define WH   128
#define UPC  64
#define INC  32
#define OUTC 64

typedef __attribute__((ext_vector_type(16))) _Float16 v16h;
typedef __attribute__((ext_vector_type(8)))  float    v8f;

union FragH { v16h v; unsigned u[8]; };

__device__ __forceinline__ v8f wmma_f16(v16h a, v16h b, v8f c) {
  // D = A(16x32 f16) * B(32x16 f16) + C(16x16 f32)
  return __builtin_amdgcn_wmma_f32_16x16x32_f16(
      /*neg_a=*/false, a, /*neg_b=*/false, b,
      /*c_mod=*/(short)0, c, /*reuse_a=*/false, /*reuse_b=*/false);
}

// A fragment: 16x32 f16 tile from LDS [m][k], row stride lda halves (even).
// Layout (ISA 7.12.2): lane L -> M = L%16; lanes0-15 K=0-7,16-23; lanes16-31 K=8-15,24-31.
__device__ __forceinline__ v16h load_A(const _Float16* s, int lda, int m0, int k0) {
  int lane = threadIdx.x & 31;
  int m  = m0 + (lane & 15);
  int kh = (lane >> 4) << 3;
  const unsigned* p = (const unsigned*)(s + m * lda + k0 + kh);
  FragH f;
#pragma unroll
  for (int j = 0; j < 4; ++j) { f.u[j] = p[j]; f.u[4 + j] = p[8 + j]; }
  return f.v;
}

// B fragment: 32x16 (KxN) built from weights stored in LDS as [n][k] f16, stride ldb halves (even).
// Layout: lane L -> N = L%16; lanes0-15 K=0-7,16-23; lanes16-31 K=8-15,24-31 (matches load_A).
__device__ __forceinline__ v16h load_B(const _Float16* s, int ldb, int n0, int k0) {
  int lane = threadIdx.x & 31;
  int n  = n0 + (lane & 15);
  int kb = k0 + ((lane >> 4) << 3);
  const unsigned* p = (const unsigned*)(s + n * ldb + kb);
  FragH f;
#pragma unroll
  for (int j = 0; j < 4; ++j) { f.u[j] = p[j]; f.u[4 + j] = p[8 + j]; }
  return f.v;
}

// C frag (f32 16x16, ISA layout: VGPR r -> M=r (+8 for hi half-wave), lane -> N) to f16 A-major LDS tile.
__device__ __forceinline__ void store_C_f16(_Float16* s, int lda, int m0, int n0, v8f c, bool relu) {
  int lane = threadIdx.x & 31;
  int n  = n0 + (lane & 15);
  int mb = (lane >> 4) << 3;
#pragma unroll
  for (int r = 0; r < 8; ++r) {
    float v = c[r];
    if (relu) v = v > 0.f ? v : 0.f;
    s[(m0 + mb + r) * lda + n] = (_Float16)v;
  }
}

// C frag to f32 channel-major LDS tile outT[n][p] for coalesced global stores.
__device__ __forceinline__ void store_C_T(float* s, int ldp, int n0, int p0, v8f c) {
  int lane = threadIdx.x & 31;
  int n  = n0 + (lane & 15);
  int mb = (lane >> 4) << 3;
#pragma unroll
  for (int r = 0; r < 8; ++r) s[n * ldp + p0 + mb + r] = c[r];
}

// ---------------- Kernel 0: zero LN accumulators ----------------
__global__ void k_zero(float* stats) {
  if (threadIdx.x < 2 * B_) stats[threadIdx.x] = 0.f;
}

// ---------------- Kernel 1: convT(2,2) + mlp1 -> identity, LN partial sums ----------------
// grid (HH, B_), block 128 (4 waves). Wave handles input row hin, 16-input-col chunks (chunk, chunk+4).
__global__ void __launch_bounds__(128) k_conv_mlp1(
    const float* __restrict__ x, const float* __restrict__ y,
    const float* __restrict__ sArr, const float* __restrict__ bArr,
    const float* __restrict__ Wup, const float* __restrict__ bup,
    const float* __restrict__ W1, const float* __restrict__ b1,
    float* __restrict__ identity, float* __restrict__ stats)
{
  extern __shared__ char smem[];
  _Float16* wupt = (_Float16*)smem;            // 4*32*66 halves: [kl][o][c]
  _Float16* w1t  = wupt + 4 * 32 * 66;         // 64*66 halves:   [n][k]
  float*    b1eff = (float*)(w1t + 64 * 66);   // 64 floats
  char* pw = (char*)(b1eff + 64);

  int tid = threadIdx.x, lane = tid & 31, wave = tid >> 5;
  _Float16* A1   = (_Float16*)(pw + wave * 27456);  // 16*66 halves: x tile [px][c]
  _Float16* A2   = A1 + 16 * 66;                    // 64*66 halves: [px][up(0..31)|y(32..63)]
  float*    outT = (float*)(A2 + 64 * 66);          // 64*66 floats: identity [ch][px]

  int hin = blockIdx.x, bb = blockIdx.y;
  float sv = sArr[bb], bv = bArr[bb];

  // Stage weights (f16) once per workgroup.
  for (int i = tid; i < 64 * 32 * 4; i += 128) {        // Wup[c][o][k][l] -> wupt[kl][o][c]
    int l = i & 1, k = (i >> 1) & 1, o = (i >> 2) & 31, c = i >> 7;
    wupt[((k * 2 + l) * 32 + o) * 66 + c] = (_Float16)Wup[i];
  }
  for (int i = tid; i < 64 * 64; i += 128) {            // W1 (64,66): keep first 64 K's
    int n = i >> 6, k = i & 63;
    w1t[n * 66 + k] = (_Float16)W1[n * 66 + k];
  }
  if (tid < 64)                                          // fold ev channels into bias
    b1eff[tid] = b1[tid] + W1[tid * 66 + 64] * sv + W1[tid * 66 + 65] * bv;
  __syncthreads();

  // Hoist mlp1 B fragments into VGPRs (reused by 4 M-tiles x 2 chunks).
  v16h w1b0[4], w1b1[4];
#pragma unroll
  for (int nb = 0; nb < 4; ++nb) {
    w1b0[nb] = load_B(w1t, 66, nb * 16, 0);
    w1b1[nb] = load_B(w1t, 66, nb * 16, 32);
  }

  float bup0 = bup[lane & 15], bup1 = bup[16 + (lane & 15)];
  float lsum = 0.f, lsq = 0.f;

  for (int chunk = wave; chunk < 8; chunk += 4) {
    int w0in = chunk * 16;
    if (chunk + 4 < 8)
      __builtin_prefetch(&x[((size_t)(bb * UPC) * HH + hin) * WH + (chunk + 4) * 16], 0, 0);

    // Stage x tile [16 input px][64 ch] as f16.
    for (int i = lane; i < 64 * 16; i += 32) {
      int c = i >> 4, p = i & 15;
      A1[p * 66 + c] = (_Float16)x[((bb * UPC + c) * HH + hin) * WH + w0in + p];
    }
    // Stage y tile into A2[:,32+c]; patch pixel p: row=2hin+(p>>5), col=2*w0in+(p&31).
    for (int i = lane; i < 32 * 64; i += 32) {
      int c = i >> 6, p = i & 63;
      int row = 2 * hin + (p >> 5), col = 2 * w0in + (p & 31);
      A2[p * 66 + 32 + c] = (_Float16)y[((bb * INC + c) * H_ + row) * W_ + col];
    }
    __syncthreads();

    v16h xa0 = load_A(A1, 66, 0, 0);
    v16h xa1 = load_A(A1, 66, 0, 32);
#pragma unroll
    for (int kl = 0; kl < 4; ++kl) {
#pragma unroll
      for (int nb = 0; nb < 2; ++nb) {
        float bc = nb ? bup1 : bup0;
        v8f acc = {bc, bc, bc, bc, bc, bc, bc, bc};
        acc = wmma_f16(xa0, load_B(wupt + kl * 32 * 66, 66, nb * 16, 0),  acc);
        acc = wmma_f16(xa1, load_B(wupt + kl * 32 * 66, 66, nb * 16, 32), acc);
        // Scatter: element (m,n) -> pixel p = (kl>>1)*32 + 2m + (kl&1), channel nb*16+n.
        int n = nb * 16 + (lane & 15);
        int mb = (lane >> 4) << 3;
#pragma unroll
        for (int r = 0; r < 8; ++r) {
          int m = mb + r;
          int p = (kl >> 1) * 32 + 2 * m + (kl & 1);
          A2[p * 66 + n] = (_Float16)acc[r];
        }
      }
    }
    __syncthreads();

    // mlp1: 4 M-tiles x 4 N-blocks, K=64 (2 chunks); B fragments already in registers.
#pragma unroll
    for (int mt = 0; mt < 4; ++mt) {
      v16h a0 = load_A(A2, 66, mt * 16, 0);
      v16h a1 = load_A(A2, 66, mt * 16, 32);
#pragma unroll
      for (int nb = 0; nb < 4; ++nb) {
        float bc = b1eff[nb * 16 + (lane & 15)];
        v8f acc = {bc, bc, bc, bc, bc, bc, bc, bc};
        acc = wmma_f16(a0, w1b0[nb], acc);
        acc = wmma_f16(a1, w1b1[nb], acc);
        store_C_T(outT, 66, nb * 16, mt * 16, acc);
      }
    }
    __syncthreads();

    // Coalesced identity store + LN partials.
    for (int i = lane; i < 64 * 64; i += 32) {
      int n = i >> 6, p = i & 63;
      float v = outT[n * 66 + p];
      int row = 2 * hin + (p >> 5), col = 2 * w0in + (p & 31);
      identity[((bb * OUTC + n) * H_ + row) * W_ + col] = v;
      lsum += v; lsq += v * v;
    }
    __syncthreads();
  }

#pragma unroll
  for (int off = 16; off > 0; off >>= 1) {
    lsum += __shfl_xor(lsum, off, 32);
    lsq  += __shfl_xor(lsq,  off, 32);
  }
  if (lane == 0) {
    atomicAdd(&stats[bb * 2],     lsum);
    atomicAdd(&stats[bb * 2 + 1], lsq);
  }
}

// ---------------- Kernel 2: finalize LN stats ----------------
__global__ void k_stats(const float* __restrict__ stats, float* __restrict__ muinv) {
  int b = threadIdx.x;
  if (b < B_) {
    float n   = (float)OUTC * H_ * W_;
    float mu  = stats[2 * b] / n;
    float var = stats[2 * b + 1] / n - mu * mu;
    muinv[2 * b]     = mu;
    muinv[2 * b + 1] = rsqrtf(var + 1e-5f);
  }
}

// ---------------- Kernel 3: LN + mlp3 (x2) + identity, in place in d_out ----------------
// grid (W_/64, H_, B_), block 128: wave handles 16-pixel M-tile of one row.
__global__ void __launch_bounds__(128) k_ln_mlp3(
    float* __restrict__ outp,
    const float* __restrict__ sArr, const float* __restrict__ bArr,
    const float* __restrict__ W3a, const float* __restrict__ b3a,
    const float* __restrict__ W3b, const float* __restrict__ b3b,
    const float* __restrict__ W3c, const float* __restrict__ b3c,
    const float* __restrict__ ln_g, const float* __restrict__ ln_b,
    const float* __restrict__ muinv)
{
  extern __shared__ char smem[];
  _Float16* w3at = (_Float16*)smem;         // 32*66
  _Float16* w3bt = w3at + 32 * 66;          // 32*34
  _Float16* w3ct = w3bt + 32 * 34;          // 64*34
  float* ba1 = (float*)(w3ct + 64 * 34);    // 32
  float* ba2 = ba1 + 32;                    // 32
  char* pw = (char*)(ba2 + 32);

  int tid = threadIdx.x, lane = tid & 31, wave = tid >> 5;
  _Float16* A    = (_Float16*)(pw + wave * 9920);   // 16*66
  _Float16* Ah   = A + 16 * 66;                     // 16*34 (h1 then h2)
  _Float16* At   = Ah + 16 * 34;                    // 16*66 (t1)
  float*    outT = (float*)(At + 16 * 66);          // 64*18

  int bb = blockIdx.z, h = blockIdx.y;
  int w0 = blockIdx.x * 64 + wave * 16;
  float sv = sArr[bb], bv = bArr[bb];
  float mu = muinv[2 * bb], inv = muinv[2 * bb + 1];

  for (int i = tid; i < 32 * 64; i += 128) { int n = i >> 6, k = i & 63; w3at[n * 66 + k] = (_Float16)W3a[n * 66 + k]; }
  for (int i = tid; i < 32 * 32; i += 128) { int n = i >> 5, k = i & 31; w3bt[n * 34 + k] = (_Float16)W3b[n * 32 + k]; }
  for (int i = tid; i < 64 * 32; i += 128) { int n = i >> 5, k = i & 31; w3ct[n * 34 + k] = (_Float16)W3c[n * 32 + k]; }
  if (tid < 32) {   // SEP=0.5: ev pairs are (0.5s, b) and (0.5s, b+0.5s); fold into biases
    float hs = 0.5f * sv;
    ba1[tid] = b3a[tid] + W3a[tid * 66 + 64] * hs + W3a[tid * 66 + 65] * bv;
    ba2[tid] = b3a[tid] + W3a[tid * 66 + 64] * hs + W3a[tid * 66 + 65] * (bv + hs);
  }
  __syncthreads();

  // Hoist all mlp3 B fragments into VGPRs (each used by both applications).
  v16h wab0[2], wab1[2], wbb[2], wcb[4];
#pragma unroll
  for (int nb = 0; nb < 2; ++nb) {
    wab0[nb] = load_B(w3at, 66, nb * 16, 0);
    wab1[nb] = load_B(w3at, 66, nb * 16, 32);
    wbb[nb]  = load_B(w3bt, 34, nb * 16, 0);
  }
#pragma unroll
  for (int nb = 0; nb < 4; ++nb) wcb[nb] = load_B(w3ct, 34, nb * 16, 0);

  // LayerNorm tile -> f16 A-major.
  for (int i = lane; i < 64 * 16; i += 32) {
    int ch = i >> 4, px = i & 15;
    int gi = ((bb * OUTC + ch) * H_ + h) * W_ + w0 + px;
    int li = (ch * H_ + h) * W_ + w0 + px;
    float v = (outp[gi] - mu) * inv * ln_g[li] + ln_b[li];
    A[px * 66 + ch] = (_Float16)v;
  }
  __syncthreads();

  v8f tacc[4];
#pragma unroll
  for (int app = 0; app < 2; ++app) {
    const _Float16* inA = (app == 0) ? A : At;
    const float*    ba  = (app == 0) ? ba1 : ba2;
    v16h a0 = load_A(inA, 66, 0, 0);
    v16h a1 = load_A(inA, 66, 0, 32);
#pragma unroll
    for (int nb = 0; nb < 2; ++nb) {   // h1 = relu(W3a @ f + ba)
      float bc = ba[nb * 16 + (lane & 15)];
      v8f acc = {bc, bc, bc, bc, bc, bc, bc, bc};
      acc = wmma_f16(a0, wab0[nb], acc);
      acc = wmma_f16(a1, wab1[nb], acc);
      store_C_f16(Ah, 34, 0, nb * 16, acc, true);
    }
    __syncthreads();
    v16h h1 = load_A(Ah, 34, 0, 0);
#pragma unroll
    for (int nb = 0; nb < 2; ++nb) {   // h2 = relu(W3b @ h1 + b3b)
      float bc = b3b[nb * 16 + (lane & 15)];
      v8f acc = {bc, bc, bc, bc, bc, bc, bc, bc};
      acc = wmma_f16(h1, wbb[nb], acc);
      store_C_f16(Ah, 34, 0, nb * 16, acc, true);
    }
    __syncthreads();
    v16h h2 = load_A(Ah, 34, 0, 0);
#pragma unroll
    for (int nb = 0; nb < 4; ++nb) {   // t = W3c @ h2 + b3c
      float bc = b3c[nb * 16 + (lane & 15)];
      v8f acc = {bc, bc, bc, bc, bc, bc, bc, bc};
      tacc[nb] = wmma_f16(h2, wcb[nb], acc);
    }
    if (app == 0) {
#pragma unroll
      for (int nb = 0; nb < 4; ++nb) store_C_f16(At, 66, 0, nb * 16, tacc[nb], false);
      __syncthreads();
    }
  }

#pragma unroll
  for (int nb = 0; nb < 4; ++nb) store_C_T(outT, 18, nb * 16, 0, tacc[nb]);
  __syncthreads();
  for (int i = lane; i < 64 * 16; i += 32) {   // out = t2 + identity (in place)
    int ch = i >> 4, px = i & 15;
    int gi = ((bb * OUTC + ch) * H_ + h) * W_ + w0 + px;
    outp[gi] = outT[ch * 18 + px] + outp[gi];
  }
}

extern "C" void kernel_launch(void* const* d_in, const int* in_sizes, int n_in,
                              void* d_out, int out_size, void* d_ws, size_t ws_size,
                              hipStream_t stream) {
  const float* x    = (const float*)d_in[0];
  const float* y    = (const float*)d_in[1];
  const float* s    = (const float*)d_in[2];
  const float* b    = (const float*)d_in[3];
  const float* Wup  = (const float*)d_in[4];
  const float* bup  = (const float*)d_in[5];
  const float* W1   = (const float*)d_in[6];
  const float* b1   = (const float*)d_in[7];
  const float* W3a  = (const float*)d_in[8];
  const float* b3a  = (const float*)d_in[9];
  const float* W3b  = (const float*)d_in[10];
  const float* b3b  = (const float*)d_in[11];
  const float* W3c  = (const float*)d_in[12];
  const float* b3c  = (const float*)d_in[13];
  const float* ln_g = (const float*)d_in[14];
  const float* ln_b = (const float*)d_in[15];
  float* out   = (float*)d_out;
  float* stats = (float*)d_ws;        // [2*B] sum/sumsq
  float* muinv = stats + 2 * B_;      // [2*B] mu/inv

  k_zero<<<1, 32, 0, stream>>>(stats);

  size_t sm1 = (4 * 32 * 66 + 64 * 66) * sizeof(_Float16) + 64 * sizeof(float)
             + 4 * ((16 * 66 + 64 * 66) * sizeof(_Float16) + 64 * 66 * sizeof(float));
  k_conv_mlp1<<<dim3(HH, B_), 128, sm1, stream>>>(x, y, s, b, Wup, bup, W1, b1, out, stats);

  k_stats<<<1, 32, 0, stream>>>(stats, muinv);

  size_t sm3 = (32 * 66 + 32 * 34 + 64 * 34) * sizeof(_Float16) + 64 * sizeof(float)
             + 4 * ((16 * 66 + 16 * 34 + 16 * 66) * sizeof(_Float16) + 64 * 18 * sizeof(float));
  k_ln_mlp3<<<dim3(W_ / 64, H_, B_), 128, sm3, stream>>>(out, s, b, W3a, b3a, W3b, b3b,
                                                         W3c, b3c, ln_g, ln_b, muinv);
}
